// TransformerDecoderLayer_8967891714183
// MI455X (gfx1250) — matrix operand order repaired and checked by reading, not compiled, not run on hardware
//
#include <hip/hip_runtime.h>
#include <hip/hip_bf16.h>
#include <math.h>

typedef __attribute__((ext_vector_type(16))) _Float16 v16h;
typedef __attribute__((ext_vector_type(8)))  _Float16 h8;
typedef __attribute__((ext_vector_type(4)))  _Float16 h4;
typedef __attribute__((ext_vector_type(8)))  float    v8f;
typedef __attribute__((ext_vector_type(4)))  float    f4;

typedef __attribute__((address_space(3))) float lds_f32;

#define N_Q 128
#define BB  2
#define SS  1024
#define CC  256
#define HH  8
#define LL  4
#define FF  2048
#define DHD 32

// ---------------------------------------------------------------------------
// WMMA GEMM:  Y[M,Nf] = act( X[M,K] @ W^T + bias [+ add] )
// W is torch nn.Linear layout (Nf, K). Requires M%64==0, Nf%64==0, K%64==0.
// Block: 128 threads (4 waves). Block tile 64(M) x 64(N), K-slab 64.
// Per K-slab: 2 A-fragments, each reused across 4 B sub-tiles -> 8 v_wmma.
// LDS layouts make every fragment exactly 2x ds_load_b128:
//   As[row][k]  (64x64 f16) : lane's A frag = 2 contiguous 8-elem K runs
//   Ws2[n][k]   (64x64 f16) : lane's B frag = 2 contiguous 8-elem K runs
// ---------------------------------------------------------------------------
__global__ __launch_bounds__(128) void wmma_gemm_kernel(
    const float* __restrict__ X, const float* __restrict__ W,
    const float* __restrict__ bias, const float* __restrict__ add,
    float* __restrict__ Y, int M, int Nf, int K, int relu)
{
    __shared__ _Float16 As[64 * 64];
    __shared__ _Float16 Ws2[64 * 64];

    const int tid  = threadIdx.x;
    const int lane = tid & 31;
    const int wave = tid >> 5;
    const int mBase = blockIdx.x * 64;
    const int nBase = blockIdx.y * 64;
    const int r    = lane & 15;
    const int half = lane >> 4;

    v8f acc0 = {}, acc1 = {}, acc2 = {}, acc3 = {};

    for (int k0 = 0; k0 < K; k0 += 64) {
        // ---- stage X tile (64x64) as f16: float4 loads, b64 stores ----
#pragma unroll
        for (int it = 0; it < 8; ++it) {
            int i = it * 128 + tid;
            int row = i >> 4, c4 = i & 15;
            const float* src = X + (size_t)(mBase + row) * K + k0 + c4 * 4;
            f4 x = *(const f4*)src;
            __builtin_prefetch(src + 64, 0, 3);   // next K-slab, WGP scope
            h4 p; p[0] = (_Float16)x[0]; p[1] = (_Float16)x[1];
                  p[2] = (_Float16)x[2]; p[3] = (_Float16)x[3];
            *(h4*)&As[row * 64 + c4 * 4] = p;
        }
        // ---- stage W tile (64 N-rows x 64 K) into Ws2[n][k] ----
#pragma unroll
        for (int it = 0; it < 8; ++it) {
            int i = it * 128 + tid;
            int n = i >> 4, c4 = i & 15;
            f4 x = *(const f4*)(W + (size_t)(nBase + n) * K + k0 + c4 * 4);
            Ws2[n * 64 + c4 * 4 + 0] = (_Float16)x[0];
            Ws2[n * 64 + c4 * 4 + 1] = (_Float16)x[1];
            Ws2[n * 64 + c4 * 4 + 2] = (_Float16)x[2];
            Ws2[n * 64 + c4 * 4 + 3] = (_Float16)x[3];
        }
        __syncthreads();

#pragma unroll
        for (int ks = 0; ks < 2; ++ks) {
            union { v16h v; h8 c[2]; } a, b;
            const int abase = (wave * 16 + r) * 64 + ks * 32;
            a.c[0] = *(const h8*)&As[abase + half * 8];
            a.c[1] = *(const h8*)&As[abase + 16 + half * 8];

            const int brow = r * 64 + ks * 32 + half * 8;
            b.c[0] = *(const h8*)&Ws2[brow +  0 * 1024];
            b.c[1] = *(const h8*)&Ws2[brow +  0 * 1024 + 16];
            acc0 = __builtin_amdgcn_wmma_f32_16x16x32_f16(false, a.v, false, b.v, (short)0, acc0, false, false);
            b.c[0] = *(const h8*)&Ws2[brow + 1 * 1024];
            b.c[1] = *(const h8*)&Ws2[brow + 1 * 1024 + 16];
            acc1 = __builtin_amdgcn_wmma_f32_16x16x32_f16(false, a.v, false, b.v, (short)0, acc1, false, false);
            b.c[0] = *(const h8*)&Ws2[brow + 2 * 1024];
            b.c[1] = *(const h8*)&Ws2[brow + 2 * 1024 + 16];
            acc2 = __builtin_amdgcn_wmma_f32_16x16x32_f16(false, a.v, false, b.v, (short)0, acc2, false, false);
            b.c[0] = *(const h8*)&Ws2[brow + 3 * 1024];
            b.c[1] = *(const h8*)&Ws2[brow + 3 * 1024 + 16];
            acc3 = __builtin_amdgcn_wmma_f32_16x16x32_f16(false, a.v, false, b.v, (short)0, acc3, false, false);
        }
        __syncthreads();
    }

    // ---- store: D VGPR i -> M=i (lanes 0-15) / M=i+8 (lanes 16-31) ----
    v8f accs[4] = {acc0, acc1, acc2, acc3};
#pragma unroll
    for (int nt = 0; nt < 4; ++nt) {
        const int col = nBase + nt * 16 + (lane & 15);
        const float bc = bias[col];
#pragma unroll
        for (int i = 0; i < 8; ++i) {
            int row = mBase + wave * 16 + i + half * 8;
            float y = accs[nt][i] + bc;
            if (add)  y += add[(size_t)row * Nf + col];
            if (relu) y = y > 0.f ? y : 0.f;
            Y[(size_t)row * Nf + col] = y;
        }
    }
}

// ---------------------------------------------------------------------------
// LayerNorm over last dim (C=256). One block of 256 threads per row.
// ---------------------------------------------------------------------------
__global__ __launch_bounds__(256) void ln_kernel(
    const float* __restrict__ X, const float* __restrict__ g,
    const float* __restrict__ b, float* __restrict__ Y, float* __restrict__ Y2)
{
    __shared__ float red[256];
    const int row = blockIdx.x, t = threadIdx.x;
    float x = X[(size_t)row * CC + t];
    red[t] = x; __syncthreads();
    for (int s = 128; s > 0; s >>= 1) { if (t < s) red[t] += red[t + s]; __syncthreads(); }
    float mean = red[0] * (1.f / CC);
    __syncthreads();
    float d = x - mean;
    red[t] = d * d; __syncthreads();
    for (int s = 128; s > 0; s >>= 1) { if (t < s) red[t] += red[t + s]; __syncthreads(); }
    float var = red[0] * (1.f / CC);
    float y = d * rsqrtf(var + 1e-5f) * g[t] + b[t];
    Y[(size_t)row * CC + t] = y;
    if (Y2) Y2[(size_t)row * CC + t] = y;
}

// ---------------------------------------------------------------------------
// Self-attention: block per (b,h); thread t = query n.
// K/V staged into LDS via CDNA5 async global->LDS copies (ASYNCcnt path).
// ---------------------------------------------------------------------------
__global__ __launch_bounds__(128) void sa_kernel(
    const float* __restrict__ Q, const float* __restrict__ Kb,
    const float* __restrict__ V, float* __restrict__ O)
{
    const int b = blockIdx.x / HH, h = blockIdx.x % HH;
    __shared__ float kS[N_Q * DHD];
    __shared__ float vS[N_Q * DHD];
    const int t = threadIdx.x;

#pragma unroll
    for (int d4 = 0; d4 < DHD / 4; ++d4) {
        lds_f32* lk = (lds_f32*)&kS[t * DHD + d4 * 4];
        lds_f32* lv = (lds_f32*)&vS[t * DHD + d4 * 4];
        const float* gk = &Kb[((size_t)t * BB + b) * CC + h * DHD + d4 * 4];
        const float* gv = &V [((size_t)t * BB + b) * CC + h * DHD + d4 * 4];
        asm volatile("global_load_async_to_lds_b128 %0, %1, off"
                     :: "v"(lk), "v"(gk) : "memory");
        asm volatile("global_load_async_to_lds_b128 %0, %1, off"
                     :: "v"(lv), "v"(gv) : "memory");
    }
    float q[DHD];
    const float scale = 0.17677669529663688f;  // 32^-0.5
    for (int d = 0; d < DHD; ++d)
        q[d] = Q[((size_t)t * BB + b) * CC + h * DHD + d] * scale;
    asm volatile("s_wait_asynccnt 0x0" ::: "memory");
    __syncthreads();

    float mx = -1e30f;
    for (int m = 0; m < N_Q; ++m) {
        float s = 0.f;
        for (int d = 0; d < DHD; ++d) s += q[d] * kS[m * DHD + d];
        mx = fmaxf(mx, s);
    }
    float z = 0.f;
    for (int m = 0; m < N_Q; ++m) {
        float s = 0.f;
        for (int d = 0; d < DHD; ++d) s += q[d] * kS[m * DHD + d];
        z += __expf(s - mx);
    }
    const float zin = 1.f / z;
    float acc[DHD];
    for (int d = 0; d < DHD; ++d) acc[d] = 0.f;
    for (int m = 0; m < N_Q; ++m) {
        float s = 0.f;
        for (int d = 0; d < DHD; ++d) s += q[d] * kS[m * DHD + d];
        float p = __expf(s - mx) * zin;
        for (int d = 0; d < DHD; ++d) acc[d] += p * vS[m * DHD + d];
    }
    for (int d = 0; d < DHD; ++d)
        O[((size_t)t * BB + b) * CC + h * DHD + d] = acc[d];
}

// ---------------------------------------------------------------------------
// lvl = tgt1 @ lvl_w^T + lvl_b   (256 x 4, tiny -> plain dots)
// ---------------------------------------------------------------------------
__global__ void lvl_kernel(const float* __restrict__ X, const float* __restrict__ Wl,
                           const float* __restrict__ bl, float* __restrict__ Y)
{
    int o = blockIdx.x * blockDim.x + threadIdx.x;
    if (o >= N_Q * BB * LL) return;
    int l = o % LL, rb = o / LL;
    float s = bl[l];
    for (int k = 0; k < CC; ++k) s += X[(size_t)rb * CC + k] * Wl[(size_t)l * CC + k];
    Y[o] = s;
}

// ---------------------------------------------------------------------------
// q_memory[n,s,b,c] = sum_l lvl[n,b,l] * memory[l,s,b,c]
// 268 MB write-once output: nontemporal stores keep it out of L2 so that
// memory/Kc/Vl stay resident for the cross-attention pass.
// ---------------------------------------------------------------------------
__global__ __launch_bounds__(256) void qmem_kernel(
    const float* __restrict__ LVL, const f4* __restrict__ MEM,
    f4* __restrict__ QM)
{
    size_t i4 = (size_t)blockIdx.x * blockDim.x + threadIdx.x;   // 16,777,216 total
    int    c4 = (int)(i4 & (CC / 4 - 1));
    size_t rb = i4 >> 6;                 // (n*S + s)*B + b
    int b = (int)(rb % BB);
    size_t ns = rb / BB;
    int s = (int)(ns % SS);
    int n = (int)(ns / SS);

    const float* lv = &LVL[(size_t)(n * BB + b) * LL];
    f4 acc = {0.f, 0.f, 0.f, 0.f};
#pragma unroll
    for (int l = 0; l < LL; ++l) {
        f4 m = MEM[(((size_t)l * SS + s) * BB + b) * (CC / 4) + c4];
        acc += lv[l] * m;
    }
    __builtin_nontemporal_store(acc, &QM[i4]);
}

// ---------------------------------------------------------------------------
// Cross-attention with the level-mixture folded in:
//   kc[n] = sum_l lvl*Kc_l, vv[n] = sum_l lvl*V_l  (Kc/V are L2-resident 8MB)
// Block per (n,b,h); 128 threads, 8 s-positions each; float4 over head dim.
// ---------------------------------------------------------------------------
__global__ __launch_bounds__(128) void ca_kernel(
    const float* __restrict__ QC, const float* __restrict__ QSE,
    const float* __restrict__ LVL, const float* __restrict__ KC,
    const float* __restrict__ VL, const float* __restrict__ KP,
    float* __restrict__ O)
{
    const int idx = blockIdx.x;
    const int h = idx % HH;
    const int b = (idx / HH) % BB;
    const int n = idx / (HH * BB);
    const int t = threadIdx.x;

    __shared__ float qcS[DHD], qseS[DHD], lvlS[LL];
    __shared__ float red[128];
    __shared__ float accAll[128 * DHD];

    if (t < DHD) {
        qcS[t]  = QC [((size_t)n * BB + b) * CC + h * DHD + t];
        qseS[t] = QSE[((size_t)n * BB + b) * CC + h * DHD + t];
    }
    if (t < LL) lvlS[t] = LVL[(size_t)(n * BB + b) * LL + t];
    __syncthreads();

    const float rs = 0.125f;  // (2*DH)^-0.5
    float myS[SS / 128];
#pragma unroll
    for (int i = 0; i < SS / 128; ++i) {
        int s = i * 128 + t;
        float dot = 0.f;
#pragma unroll
        for (int d4 = 0; d4 < DHD / 4; ++d4) {
            f4 kc = {0.f, 0.f, 0.f, 0.f};
#pragma unroll
            for (int l = 0; l < LL; ++l)
                kc += lvlS[l] * *(const f4*)&KC[(((size_t)l * SS + s) * BB + b) * CC + h * DHD + d4 * 4];
            f4 qc = *(const f4*)&qcS[d4 * 4];
            f4 qs = *(const f4*)&qseS[d4 * 4];
            f4 kp = *(const f4*)&KP[((size_t)s * BB + b) * CC + h * DHD + d4 * 4];
            f4 contrib = qc * kc + qs * kp;
            dot += contrib[0] + contrib[1] + contrib[2] + contrib[3];
        }
        myS[i] = dot * rs;
    }

    float lmax = -1e30f;
#pragma unroll
    for (int i = 0; i < SS / 128; ++i) lmax = fmaxf(lmax, myS[i]);
    red[t] = lmax; __syncthreads();
    for (int s = 64; s > 0; s >>= 1) { if (t < s) red[t] = fmaxf(red[t], red[t + s]); __syncthreads(); }
    const float mx = red[0]; __syncthreads();

    float lz = 0.f;
#pragma unroll
    for (int i = 0; i < SS / 128; ++i) { myS[i] = __expf(myS[i] - mx); lz += myS[i]; }
    red[t] = lz; __syncthreads();
    for (int s = 64; s > 0; s >>= 1) { if (t < s) red[t] += red[t + s]; __syncthreads(); }
    const float zin = 1.f / red[0];

    f4 acc[DHD / 4];
#pragma unroll
    for (int d4 = 0; d4 < DHD / 4; ++d4) acc[d4] = (f4){0.f, 0.f, 0.f, 0.f};
#pragma unroll
    for (int i = 0; i < SS / 128; ++i) {
        int s = i * 128 + t;
        float p = myS[i] * zin;
#pragma unroll
        for (int d4 = 0; d4 < DHD / 4; ++d4) {
            f4 vv = {0.f, 0.f, 0.f, 0.f};
#pragma unroll
            for (int l = 0; l < LL; ++l)
                vv += lvlS[l] * *(const f4*)&VL[(((size_t)l * SS + s) * BB + b) * CC + h * DHD + d4 * 4];
            acc[d4] += p * vv;
        }
    }
#pragma unroll
    for (int d4 = 0; d4 < DHD / 4; ++d4)
        *(f4*)&accAll[t * DHD + d4 * 4] = acc[d4];
    __syncthreads();
    if (t < DHD) {
        float s = 0.f;
        for (int w = 0; w < 128; ++w) s += accAll[w * DHD + t];
        O[((size_t)n * BB + b) * CC + h * DHD + t] = s;
    }
}

// ---------------------------------------------------------------------------
extern "C" void kernel_launch(void* const* d_in, const int* in_sizes, int n_in,
                              void* d_out, int out_size, void* d_ws, size_t ws_size,
                              hipStream_t stream)
{
    (void)in_sizes; (void)n_in; (void)out_size; (void)ws_size;

    const float* tgt   = (const float*)d_in[0];
    const float* mem   = (const float*)d_in[1];
    const float* pos   = (const float*)d_in[2];   // level 0 slice = first S*B*C
    const float* qpos  = (const float*)d_in[3];
    const float* qsine = (const float*)d_in[4];
    const float* P[36];
    for (int i = 0; i < 36; ++i) P[i] = (const float*)d_in[5 + i];
    // 0 sa_qc_w 1 sa_qc_b 2 sa_qp 4 sa_kc 6 sa_kp 8 sa_v 10 sa_out
    // 12 ca_qc 14 ca_kc 16 ca_v 18 ca_kp 20 ca_qse 22 ca_out
    // 24 lvl 26 l1 28 l2 30 n1_g 31 n1_b 32 n2_g 33 n2_b 34 n3_g 35 n3_b

    const int NB = N_Q * BB;                 // 256 rows
    float* w = (float*)d_ws;
    float* t0   = w; w += NB * CC;
    float* qb   = w; w += NB * CC;
    float* kb   = w; w += NB * CC;
    float* vb   = w; w += NB * CC;
    float* sab  = w; w += NB * CC;
    float* pre1 = w; w += NB * CC;
    float* tgt1 = w; w += NB * CC;
    float* lvlb = w; w += NB * LL;
    float* Kc   = w; w += (size_t)LL * SS * BB * CC;
    float* Vl   = w; w += (size_t)LL * SS * BB * CC;
    float* kpos = w; w += (size_t)SS * BB * CC;
    float* qcb  = w; w += NB * CC;
    float* qseb = w; w += NB * CC;
    float* cab  = w; w += NB * CC;
    float* pre2 = w; w += NB * CC;
    float* tgt2 = w; w += NB * CC;
    float* ffn1 = w; w += (size_t)NB * FF;
    float* pre3 = w; w += NB * CC;

    float* out_tgt  = (float*)d_out;
    float* out_temp = out_tgt + NB * CC;
    float* out_qm   = out_temp + NB * CC;

    auto gemm = [&](const float* X, const float* Wt, const float* bias,
                    const float* add, float* Y, int M, int Nf, int K, int relu) {
        dim3 g(M / 64, Nf / 64);
        hipLaunchKernelGGL(wmma_gemm_kernel, g, dim3(128), 0, stream,
                           X, Wt, bias, add, Y, M, Nf, K, relu);
    };

    // ---- self-attention projections ----
    gemm(tgt,  P[0], P[1], nullptr, t0, NB, CC, CC, 0);       // sa_qc
    gemm(qpos, P[2], P[3], t0,      qb, NB, CC, CC, 0);       // + sa_qp
    gemm(tgt,  P[4], P[5], nullptr, t0, NB, CC, CC, 0);       // sa_kc
    gemm(qpos, P[6], P[7], t0,      kb, NB, CC, CC, 0);       // + sa_kp
    gemm(tgt,  P[8], P[9], nullptr, vb, NB, CC, CC, 0);       // sa_v
    hipLaunchKernelGGL(sa_kernel, dim3(BB * HH), dim3(128), 0, stream, qb, kb, vb, sab);
    gemm(sab, P[10], P[11], tgt, pre1, NB, CC, CC, 0);        // sa_out + residual
    hipLaunchKernelGGL(ln_kernel, dim3(NB), dim3(CC), 0, stream,
                       pre1, P[30], P[31], tgt1, (float*)nullptr);

    // ---- level weights + q_memory output (pure streaming, 268 MB write) ----
    hipLaunchKernelGGL(lvl_kernel, dim3(4), dim3(256), 0, stream, tgt1, P[24], P[25], lvlb);
    {
        size_t nq4 = (size_t)N_Q * SS * BB * (CC / 4);        // 16,777,216
        hipLaunchKernelGGL(qmem_kernel, dim3((unsigned)(nq4 / 256)), dim3(256), 0, stream,
                           lvlb, (const f4*)mem, (f4*)out_qm);
    }

    // ---- per-level projections (replaces the 34-GFLOP per-query GEMMs) ----
    gemm(mem, P[14], P[15], nullptr, Kc, LL * SS * BB, CC, CC, 0);   // ca_kc per level
    gemm(mem, P[16], P[17], nullptr, Vl, LL * SS * BB, CC, CC, 0);   // ca_v per level
    gemm(pos, P[18], P[19], nullptr, kpos, SS * BB, CC, CC, 0);      // ca_kp on pos[0]
    gemm(tgt1,  P[12], P[13], nullptr, qcb,  NB, CC, CC, 0);         // ca_qc
    gemm(qsine, P[20], P[21], nullptr, qseb, NB, CC, CC, 0);         // ca_qse

    // ---- cross-attention with folded level mixture ----
    hipLaunchKernelGGL(ca_kernel, dim3(N_Q * BB * HH), dim3(128), 0, stream,
                       qcb, qseb, lvlb, Kc, Vl, kpos, cab);
    gemm(cab, P[22], P[23], tgt1, pre2, NB, CC, CC, 0);       // ca_out + residual
    hipLaunchKernelGGL(ln_kernel, dim3(NB), dim3(CC), 0, stream,
                       pre2, P[32], P[33], tgt2, out_temp);   // tgt_temp output

    // ---- FFN ----
    gemm(tgt2, P[26], P[27], nullptr, ffn1, NB, FF, CC, 1);   // relu(l1)
    gemm(ffn1, P[28], P[29], tgt2,    pre3, NB, CC, FF, 0);   // l2 + residual
    hipLaunchKernelGGL(ln_kernel, dim3(NB), dim3(CC), 0, stream,
                       pre3, P[34], P[35], out_tgt, (float*)nullptr);
}
